// KNNGNN_1846835938186
// MI455X (gfx1250) — compile-verified, matrix operand order checked by reading, not run on hardware
//
#include <hip/hip_runtime.h>
#include <hip/hip_bf16.h>
#include <math.h>

// ---- types for WMMA ----
typedef __attribute__((ext_vector_type(16))) __bf16 v16bf;
typedef __attribute__((ext_vector_type(8)))  __bf16 v8bf;
typedef __attribute__((ext_vector_type(8)))  float  v8f;

#define NNODES 50000
#define NFEAT  256
#define NHID   128
#define NCLASS 16
#define NTRAIN 5000
#define NTEST  5000
#define KSEL   20
#define NPADM  5008   // 313 * 16  (test rows / GEMM M padding)
#define NPADN  5056   // 79 * 64   (train rows / GEMM N padding, 64-aligned)

// ---------------- elementwise / conversion kernels ----------------

__global__ void knn_f32_to_bf16(const float* __restrict__ in, __bf16* __restrict__ out,
                                size_t n) {
  size_t i = (size_t)blockIdx.x * blockDim.x + threadIdx.x;
  if (i < n) out[i] = (__bf16)in[i];
}

// out[c * rows + r] = (bf16) in[r * cols + c]   (W -> W^T in bf16)
__global__ void knn_transpose_bf16(const float* __restrict__ in, __bf16* __restrict__ out,
                                   int rows, int cols) {
  int i = blockIdx.x * blockDim.x + threadIdx.x;
  if (i >= rows * cols) return;
  int r = i / cols, c = i % cols;
  out[(size_t)c * rows + r] = (__bf16)in[i];
}

__global__ void knn_zero_f32(float* __restrict__ p, size_t n) {
  size_t i = (size_t)blockIdx.x * blockDim.x + threadIdx.x;
  size_t stride = (size_t)gridDim.x * blockDim.x;
  for (; i < n; i += stride) p[i] = 0.0f;
}

// h1 = bf16(relu(t + b1)), t is [NNODES, NHID] flat
__global__ void knn_bias_relu_bf16(const float* __restrict__ t, const float* __restrict__ b,
                                   __bf16* __restrict__ out, size_t n) {
  size_t i = (size_t)blockIdx.x * blockDim.x + threadIdx.x;
  if (i >= n) return;
  float v = t[i] + b[(int)(i & (NHID - 1))];
  out[i] = (__bf16)(v > 0.0f ? v : 0.0f);
}

// gather emb[idx[row]] + b2 -> bf16 row-major [rows, NHID]; pad rows -> 0
__global__ void knn_gather_bias_bf16(const float* __restrict__ emb, const float* __restrict__ b,
                                     const int* __restrict__ idx, int count,
                                     __bf16* __restrict__ out) {
  int row = blockIdx.x;
  int c = threadIdx.x;           // 0 .. 127
  float v = 0.0f;
  if (row < count) v = emb[(size_t)idx[row] * NHID + c] + b[c];
  out[(size_t)row * NHID + c] = (__bf16)v;
}

// ---------------- edge scatter: out[dst] += w * h[src] ----------------
// one wave per edge; lane handles 4 consecutive floats of the 128-wide row
__global__ void knn_propagate(const float* __restrict__ h, const int* __restrict__ src,
                              const int* __restrict__ dst, const float* __restrict__ w,
                              float* __restrict__ out, int nedges) {
  int wid  = (blockIdx.x * blockDim.x + threadIdx.x) >> 5;
  int lane = threadIdx.x & 31;
  if (wid >= nedges) return;
  int s = src[wid];
  int d = dst[wid];
  float wt = w[wid];
  const float4* hs = (const float4*)(h + (size_t)s * NHID);
  float* od = out + (size_t)d * NHID + lane * 4;
  float4 v = hs[lane];
  atomicAdd(od + 0, v.x * wt);
  atomicAdd(od + 1, v.y * wt);
  atomicAdd(od + 2, v.z * wt);
  atomicAdd(od + 3, v.w * wt);
}

// ---------------- WMMA GEMM: C[MxN] = A[MxK] * B^T  (B given as N x K rows) ----
// bf16 in / f32 accum. One wave computes a 16x64 strip (4 tiles), reusing the
// A fragment across 4 WMMAs per k-step. M mult of 16, N mult of 64, K mult of 32.

// 16-bit A-matrix 16x32 layout: lanes 0-15 hold row M=lane, K=[0..7]+[16..23];
// lanes 16-31 hold row M=lane-16, K=[8..15]+[24..31]. B^T rows load identically.
__device__ __forceinline__ v16bf knn_load_frag(const __bf16* p) {
  v8bf lo = *(const v8bf*)(p);
  v8bf hi = *(const v8bf*)(p + 16);
  v16bf f;
#pragma unroll
  for (int i = 0; i < 8; ++i) { f[i] = lo[i]; f[i + 8] = hi[i]; }
  return f;
}

__global__ void knn_wmma_gemm4(const __bf16* __restrict__ A, const __bf16* __restrict__ B,
                               float* __restrict__ C, int M, int N, int K, int ldc) {
  int wave = (blockIdx.x * blockDim.x + threadIdx.x) >> 5;
  int lane = threadIdx.x & 31;
  int tiles_m  = M >> 4;
  int strips_n = N >> 6;
  if (wave >= tiles_m * strips_n) return;
  int tm = wave / strips_n;
  int sn = wave - tm * strips_n;

  int half = lane >> 4;
  int r    = lane & 15;
  const __bf16* arow = A + (size_t)(tm * 16 + r) * K + half * 8;
  const __bf16* br0  = B + (size_t)((sn * 4 + 0) * 16 + r) * K + half * 8;
  const __bf16* br1  = B + (size_t)((sn * 4 + 1) * 16 + r) * K + half * 8;
  const __bf16* br2  = B + (size_t)((sn * 4 + 2) * 16 + r) * K + half * 8;
  const __bf16* br3  = B + (size_t)((sn * 4 + 3) * 16 + r) * K + half * 8;

  v8f acc0 = {}, acc1 = {}, acc2 = {}, acc3 = {};
  for (int k0 = 0; k0 < K; k0 += 32) {
    v16bf a  = knn_load_frag(arow + k0);
    v16bf b0 = knn_load_frag(br0 + k0);
    v16bf b1 = knn_load_frag(br1 + k0);
    v16bf b2 = knn_load_frag(br2 + k0);
    v16bf b3 = knn_load_frag(br3 + k0);
    acc0 = __builtin_amdgcn_wmma_f32_16x16x32_bf16(false, a, false, b0, (short)0, acc0, false, false);
    acc1 = __builtin_amdgcn_wmma_f32_16x16x32_bf16(false, a, false, b1, (short)0, acc1, false, false);
    acc2 = __builtin_amdgcn_wmma_f32_16x16x32_bf16(false, a, false, b2, (short)0, acc2, false, false);
    acc3 = __builtin_amdgcn_wmma_f32_16x16x32_bf16(false, a, false, b3, (short)0, acc3, false, false);
  }

  // C/D layout: VGPR v -> row = v + half*8, col = lane&15
  float* cbase = C + (size_t)(tm * 16 + half * 8) * ldc + (size_t)sn * 64 + r;
#pragma unroll
  for (int v = 0; v < 8; ++v) {
    size_t o = (size_t)v * ldc;
    cbase[o +  0] = acc0[v];
    cbase[o + 16] = acc1[v];
    cbase[o + 32] = acc2[v];
    cbase[o + 48] = acc3[v];
  }
}

// ---------------- fused top-K + softmax + onehot matvec ----------------
// one wave per test row
__global__ void knn_topk_softmax(const float* __restrict__ score, int lds,
                                 const int* __restrict__ idx_train,
                                 const float* __restrict__ onehot,  // [NNODES, 16]
                                 float* __restrict__ preds, int ntest, int ntrain) {
  int wid  = (blockIdx.x * blockDim.x + threadIdx.x) >> 5;
  int lane = threadIdx.x & 31;
  if (wid >= ntest) return;
  const float* row = score + (size_t)wid * lds;

  // per-lane sorted (descending) top-KSEL list, fully register resident
  float vals[KSEL];
  int   cols[KSEL];
#pragma unroll
  for (int i = 0; i < KSEL; ++i) { vals[i] = -INFINITY; cols[i] = 0; }

  for (int j = lane; j < ntrain; j += 32) {
    float v = row[j];
    if (v > vals[KSEL - 1]) {
      vals[KSEL - 1] = v;
      cols[KSEL - 1] = j;
#pragma unroll
      for (int t = KSEL - 1; t > 0; --t) {
        if (vals[t] > vals[t - 1]) {
          float tv = vals[t]; vals[t] = vals[t - 1]; vals[t - 1] = tv;
          int   tc = cols[t]; cols[t] = cols[t - 1]; cols[t - 1] = tc;
        }
      }
    }
  }

  // merge across lanes: 20 rounds of wave argmax, fused softmax + class accum
  float acc  = 0.0f;  // lanes 0..15: class accumulator (class == lane)
  float sum  = 0.0f;
  float vmax = 0.0f;
#pragma unroll 1
  for (int sel = 0; sel < KSEL; ++sel) {
    float bv = vals[0];
    int   bj = cols[0];
    int   bl = lane;
#pragma unroll
    for (int off = 16; off; off >>= 1) {
      float ov = __shfl_xor(bv, off, 32);
      int   oj = __shfl_xor(bj, off, 32);
      int   ol = __shfl_xor(bl, off, 32);
      if (ov > bv || (ov == bv && ol < bl)) { bv = ov; bj = oj; bl = ol; }
    }
    if (lane == bl) {  // winner pops its head (static-index shift)
#pragma unroll
      for (int t = 0; t < KSEL - 1; ++t) { vals[t] = vals[t + 1]; cols[t] = cols[t + 1]; }
      vals[KSEL - 1] = -INFINITY;
    }
    if (sel == 0) vmax = bv;
    float e = __expf(bv - vmax);
    sum += e;
    if (lane < NCLASS) acc += e * onehot[(size_t)idx_train[bj] * NCLASS + lane];
  }
  if (lane < NCLASS) preds[(size_t)wid * NCLASS + lane] = acc / sum;
}

// ---------------- host side ----------------

extern "C" void kernel_launch(void* const* d_in, const int* in_sizes, int n_in,
                              void* d_out, int out_size, void* d_ws, size_t ws_size,
                              hipStream_t stream) {
  (void)n_in; (void)out_size; (void)ws_size;

  const float* features = (const float*)d_in[0];
  const float* edge_w   = (const float*)d_in[1];
  const float* onehot   = (const float*)d_in[2];
  const float* W1       = (const float*)d_in[3];
  const float* b1       = (const float*)d_in[4];
  const float* W2       = (const float*)d_in[5];
  const float* b2       = (const float*)d_in[6];
  const int*   eidx     = (const int*)d_in[7];
  const int*   idx_tr   = (const int*)d_in[8];
  const int*   idx_te   = (const int*)d_in[9];
  float* preds = (float*)d_out;

  const int E = in_sizes[7] / 2;
  const int* esrc = eidx;
  const int* edst = eidx + E;

  // workspace carve-out (256B aligned)
  char* ws = (char*)d_ws;
  size_t off = 0;
  auto carve = [&](size_t bytes) -> char* {
    char* p = ws + off;
    off = (off + bytes + 255) & ~(size_t)255;
    return p;
  };
  __bf16* Xb  = (__bf16*)carve((size_t)NNODES * NFEAT * 2);
  __bf16* W1T = (__bf16*)carve((size_t)NHID * NFEAT * 2);
  __bf16* W2T = (__bf16*)carve((size_t)NHID * NHID * 2);
  float*  T1  = (float*)carve((size_t)NNODES * NHID * 4);   // XW1, later h1@W2
  float*  T2  = (float*)carve((size_t)NNODES * NHID * 4);   // propagate dest 1
  __bf16* H1b = (__bf16*)carve((size_t)NNODES * NHID * 2);
  float*  EMB = (float*)carve((size_t)NNODES * NHID * 4);
  __bf16* TEb = (__bf16*)carve((size_t)NPADM * NHID * 2);
  __bf16* TRb = (__bf16*)carve((size_t)NPADN * NHID * 2);
  float*  SC  = (float*)carve((size_t)NPADM * NPADN * 4);

  // 1) convert operands to bf16 (weights transposed so B tiles load as rows)
  {
    size_t n = (size_t)NNODES * NFEAT;
    knn_f32_to_bf16<<<(unsigned)((n + 255) / 256), 256, 0, stream>>>(features, Xb, n);
  }
  knn_transpose_bf16<<<(NFEAT * NHID + 255) / 256, 256, 0, stream>>>(W1, W1T, NFEAT, NHID);
  knn_transpose_bf16<<<(NHID * NHID + 255) / 256, 256, 0, stream>>>(W2, W2T, NHID, NHID);

  // 2) T1 = X @ W1   (M=50000, N=128, K=256); 16x64 strips -> 2 strips per tile-row
  {
    int waves = (NNODES / 16) * (NHID / 64);
    knn_wmma_gemm4<<<(waves + 7) / 8, 256, 0, stream>>>(Xb, W1T, T1, NNODES, NHID, NFEAT, NHID);
  }

  // 3) T2 = propagate(T1)
  knn_zero_f32<<<2048, 256, 0, stream>>>(T2, (size_t)NNODES * NHID);
  knn_propagate<<<(E + 7) / 8, 256, 0, stream>>>(T1, esrc, edst, edge_w, T2, E);

  // 4) H1b = bf16(relu(T2 + b1))
  {
    size_t n = (size_t)NNODES * NHID;
    knn_bias_relu_bf16<<<(unsigned)((n + 255) / 256), 256, 0, stream>>>(T2, b1, H1b, n);
  }

  // 5) T1 = H1 @ W2   (M=50000, N=128, K=128)
  {
    int waves = (NNODES / 16) * (NHID / 64);
    knn_wmma_gemm4<<<(waves + 7) / 8, 256, 0, stream>>>(H1b, W2T, T1, NNODES, NHID, NHID, NHID);
  }

  // 6) EMB = propagate(T1)   (+ b2 folded into the gathers below)
  knn_zero_f32<<<2048, 256, 0, stream>>>(EMB, (size_t)NNODES * NHID);
  knn_propagate<<<(E + 7) / 8, 256, 0, stream>>>(T1, esrc, edst, edge_w, EMB, E);

  // 7) gather test/train embeddings (+b2) -> bf16, zero-padded
  knn_gather_bias_bf16<<<NPADM, NHID, 0, stream>>>(EMB, b2, idx_te, NTEST, TEb);
  knn_gather_bias_bf16<<<NPADN, NHID, 0, stream>>>(EMB, b2, idx_tr, NTRAIN, TRb);

  // 8) SC = emb_test @ emb_train^T   (M=5008, N=5056, K=128)
  {
    int waves = (NPADM / 16) * (NPADN / 64);
    knn_wmma_gemm4<<<(waves + 7) / 8, 256, 0, stream>>>(TEb, TRb, SC, NPADM, NPADN, NHID, NPADN);
  }

  // 9) fused top-K(20) + softmax + onehot matvec -> preds [5000, 16]
  knn_topk_softmax<<<(NTEST + 7) / 8, 256, 0, stream>>>(SC, NPADN, idx_tr, onehot,
                                                        preds, NTEST, NTRAIN);
}
